// VisionTransformer_52364241273509
// MI455X (gfx1250) — matrix operand order repaired and checked by reading, not compiled, not run on hardware
//
#include <hip/hip_runtime.h>

#define DIM    768
#define FF     3072
#define NL     12
#define NLM    2
#define NE     8
#define NH     12
#define HD     64
#define BATCH  4
#define SEQ    129
#define TOK    (SEQ*BATCH)   /* 516 */
#define MPAD   640           /* TOK padded to 128 multiple */
#define DOUT   512
#define NPATCH 512           /* 4 images * 128 patches */

typedef __bf16 bf16;
typedef __attribute__((ext_vector_type(16))) __bf16 v16bf;
typedef __attribute__((ext_vector_type(8)))  float  v8f;

static __device__ __forceinline__ bf16 f2b(float f) {
  unsigned u = __builtin_bit_cast(unsigned, f);
  u = (u + 0x7FFFu + ((u >> 16) & 1u)) >> 16;           // RNE
  unsigned short s = (unsigned short)u;
  return __builtin_bit_cast(bf16, s);
}

// ---------------------------------------------------------------------------
// C[M,N] = act(A[Mpad,K]bf16 @ W[N,K]bf16^T + bias) (+res). f32 WMMA accum.
// Block 256 thr (8 waves), tile 128x128, wave tile 32x64 (2x4 WMMA accs).
// Tiles stream global->LDS via async-DMA (double buffered), stored in LDS in
// exact WMMA fragment order so each fragment is one 32B contiguous LDS load.
// Requires: M padded rows zero-filled, N %128==0, K %32==0, 16B alignment.
// ---------------------------------------------------------------------------
__global__ __launch_bounds__(256) void gemm_wmma(
    const bf16* __restrict__ A, const bf16* __restrict__ W,
    const float* __restrict__ bias, const float* __restrict__ res,
    float* __restrict__ C, bf16* __restrict__ Cb,
    int M, int N, int K, int act)
{
  __shared__ bf16 As[2][4096];   // [buf][subtile*512 + lane*16 .. +15]
  __shared__ bf16 Bs[2][4096];

  const int tid  = threadIdx.x;
  const int m0   = (int)blockIdx.y * 128;
  const int n0   = (int)blockIdx.x * 128;
  const int lane = tid & 31;
  const int wv   = tid >> 5;
  const int wm   = wv & 3;          // 4 wave rows  (32 M each)
  const int wn   = wv >> 2;         // 2 wave cols  (64 N each)
  const int mrow = lane & 15;
  const int half = lane >> 4;

  v8f acc[2][4];
#pragma unroll
  for (int ti = 0; ti < 2; ++ti)
#pragma unroll
    for (int tj = 0; tj < 4; ++tj)
#pragma unroll
      for (int r = 0; r < 8; ++r) acc[ti][tj][r] = 0.0f;

  // Per-thread copy plan: 512 16B-chunks per 128x32 tile, 2 per thread.
  // Chunk c: row r=c>>2, k-offset kc=(c&3)*8 (bf16). LDS dest = fragment order.
  unsigned baseA = (unsigned)(size_t)(void*)&As[0][0];
  unsigned baseB = (unsigned)(size_t)(void*)&Bs[0][0];
  unsigned ldsA[2], ldsB[2];
  size_t   gA[2], gB[2];
#pragma unroll
  for (int j = 0; j < 2; ++j) {
    int c = tid + j * 256;
    int r = c >> 2, kc = (c & 3) * 8;
    gA[j] = (size_t)(m0 + r) * K + kc;
    gB[j] = (size_t)(n0 + r) * K + kc;
    // A frag: lane=(m%16)+16*((k>>3)&1), 16B part = k>>4
    ldsA[j] = (unsigned)(((r >> 4) * 512 + ((r & 15) + 16 * ((kc >> 3) & 1)) * 16 + (kc >> 4) * 8) * 2);
    // B frag: lane=(n%16)+16*((k>>3)&1), 16B part = k>>4
    ldsB[j] = (unsigned)(((r >> 4) * 512 + ((r & 15) + 16 * ((kc >> 3) & 1)) * 16 + (kc >> 4) * 8) * 2);
  }

  auto issue = [&](int buf, int k0) {
    unsigned bufOfs = (unsigned)buf * 8192u;
#pragma unroll
    for (int j = 0; j < 2; ++j) {
      unsigned la = baseA + bufOfs + ldsA[j];
      unsigned long long ga = (unsigned long long)(size_t)(A + gA[j] + k0);
      asm volatile("global_load_async_to_lds_b128 %0, %1, off"
                   :: "v"(la), "v"(ga) : "memory");
      unsigned lb = baseB + bufOfs + ldsB[j];
      unsigned long long gb = (unsigned long long)(size_t)(W + gB[j] + k0);
      asm volatile("global_load_async_to_lds_b128 %0, %1, off"
                   :: "v"(lb), "v"(gb) : "memory");
    }
  };

  issue(0, 0);
  asm volatile("s_wait_asynccnt 0" ::: "memory");
  __syncthreads();

  int buf = 0;
  for (int k0 = 0; k0 < K; k0 += 32) {
    if (k0 + 32 < K) issue(buf ^ 1, k0 + 32);   // overlap copy with compute

    v16bf af[2], bfr[4];
#pragma unroll
    for (int ti = 0; ti < 2; ++ti)
      af[ti] = *(const v16bf*)&As[buf][(wm * 2 + ti) * 512 + lane * 16];
#pragma unroll
    for (int tj = 0; tj < 4; ++tj)
      bfr[tj] = *(const v16bf*)&Bs[buf][(wn * 4 + tj) * 512 + lane * 16];

#pragma unroll
    for (int ti = 0; ti < 2; ++ti)
#pragma unroll
      for (int tj = 0; tj < 4; ++tj)
        acc[ti][tj] = __builtin_amdgcn_wmma_f32_16x16x32_bf16(
            false, af[ti], false, bfr[tj], (short)0, acc[ti][tj], false, false);

    asm volatile("s_wait_asynccnt 0" ::: "memory");
    __syncthreads();
    buf ^= 1;
  }

  // epilogue: bias / qgelu / residual ; optional secondary bf16 output
#pragma unroll
  for (int ti = 0; ti < 2; ++ti) {
#pragma unroll
    for (int tj = 0; tj < 4; ++tj) {
      int gn = n0 + wn * 64 + tj * 16 + mrow;
#pragma unroll
      for (int r = 0; r < 8; ++r) {
        int gm = m0 + wm * 32 + ti * 16 + r + 8 * half;
        if (gm < M) {
          float v = acc[ti][tj][r];
          if (bias) v += bias[gn];
          if (act == 1) v = v * (1.0f / (1.0f + __expf(-1.702f * v)));
          if (res) v += res[(size_t)gm * N + gn];
          if (C)  C[(size_t)gm * N + gn] = v;
          if (Cb) Cb[(size_t)gm * N + gn] = f2b(v);
        }
      }
    }
  }
}

// ---------------------------------------------------------------------------
// LayerNorm over last dim (=768); optional f32 and bf16 outputs.
// ---------------------------------------------------------------------------
__global__ __launch_bounds__(256) void ln_kernel(
    const float* __restrict__ X, float* __restrict__ Y, bf16* __restrict__ Y16,
    const float* __restrict__ w, const float* __restrict__ b)
{
  __shared__ float red[256];
  const int t = blockIdx.x, tid = threadIdx.x;
  float v[3];
#pragma unroll
  for (int j = 0; j < 3; ++j) v[j] = X[(size_t)t * DIM + tid + j * 256];

  float s = v[0] + v[1] + v[2];
  red[tid] = s; __syncthreads();
  for (int o = 128; o > 0; o >>= 1) { if (tid < o) red[tid] += red[tid + o]; __syncthreads(); }
  float mean = red[0] * (1.0f / DIM);
  __syncthreads();

  float sq = 0.f;
#pragma unroll
  for (int j = 0; j < 3; ++j) { float d = v[j] - mean; sq += d * d; }
  red[tid] = sq; __syncthreads();
  for (int o = 128; o > 0; o >>= 1) { if (tid < o) red[tid] += red[tid + o]; __syncthreads(); }
  float rstd = rsqrtf(red[0] * (1.0f / DIM) + 1e-5f);

#pragma unroll
  for (int j = 0; j < 3; ++j) {
    int d = tid + j * 256;
    float y = (v[j] - mean) * rstd * w[d] + b[d];
    if (Y)   Y[(size_t)t * DIM + d] = y;
    if (Y16) Y16[(size_t)t * DIM + d] = f2b(y);
  }
}

// ---------------------------------------------------------------------------
// Attention (hd=64): one block per (b,h,q), 64 threads. bf16 output.
// ---------------------------------------------------------------------------
__global__ __launch_bounds__(64) void attn_kernel(
    const float* __restrict__ qkv, bf16* __restrict__ O16)
{
  __shared__ float qs[64];
  __shared__ float sc[160];
  __shared__ float red[64];
  const int q = blockIdx.x % SEQ;
  const int h = (blockIdx.x / SEQ) % NH;
  const int b = blockIdx.x / (SEQ * NH);
  const int tid = threadIdx.x;
  const int tq = q * BATCH + b;

  qs[tid] = qkv[(size_t)tq * 3 * DIM + h * HD + tid] * 0.125f;
  for (int k = tid; k < 160; k += 64) sc[k] = -1e30f;
  __syncthreads();

  for (int k = tid; k < SEQ; k += 64) {
    const float* kp = qkv + (size_t)(k * BATCH + b) * 3 * DIM + DIM + h * HD;
    float dot = 0.f;
#pragma unroll
    for (int d = 0; d < HD; ++d) dot += qs[d] * kp[d];
    sc[k] = dot;
  }
  __syncthreads();

  float lm = -1e30f;
  for (int k = tid; k < 160; k += 64) lm = fmaxf(lm, sc[k]);
  red[tid] = lm; __syncthreads();
  for (int o = 32; o > 0; o >>= 1) { if (tid < o) red[tid] = fmaxf(red[tid], red[tid + o]); __syncthreads(); }
  float mx = red[0];
  __syncthreads();

  float ls = 0.f;
  for (int k = tid; k < SEQ; k += 64) { float e = __expf(sc[k] - mx); sc[k] = e; ls += e; }
  red[tid] = ls; __syncthreads();
  for (int o = 32; o > 0; o >>= 1) { if (tid < o) red[tid] += red[tid + o]; __syncthreads(); }
  float inv = 1.0f / red[0];

  float acc = 0.f;
  for (int k = 0; k < SEQ; ++k) {
    const float* vp = qkv + (size_t)(k * BATCH + b) * 3 * DIM + 2 * DIM + h * HD;
    acc += sc[k] * vp[tid];
  }
  O16[(size_t)tq * DIM + h * HD + tid] = f2b(acc * inv);
}

// ---------------------------------------------------------------------------
// Router: logits, softmax, top-2 renormalized combine weights.
// ---------------------------------------------------------------------------
__global__ __launch_bounds__(256) void router_kernel(
    const float* __restrict__ Hm, const float* __restrict__ gate,
    float* __restrict__ comb, float* __restrict__ logits_out)
{
  __shared__ float part[NE][256];
  const int t = blockIdx.x, tid = threadIdx.x;
  float hv[3];
#pragma unroll
  for (int j = 0; j < 3; ++j) hv[j] = Hm[(size_t)t * DIM + tid + j * 256];
#pragma unroll
  for (int e = 0; e < NE; ++e) {
    float p = 0.f;
#pragma unroll
    for (int j = 0; j < 3; ++j) p += hv[j] * gate[(size_t)e * DIM + tid + j * 256];
    part[e][tid] = p;
  }
  __syncthreads();
  for (int o = 128; o > 0; o >>= 1) {
    if (tid < o)
#pragma unroll
      for (int e = 0; e < NE; ++e) part[e][tid] += part[e][tid + o];
    __syncthreads();
  }
  if (tid == 0) {
    float lg[NE], w[NE], mx = -1e30f, sum = 0.f;
#pragma unroll
    for (int e = 0; e < NE; ++e) { lg[e] = part[e][0]; logits_out[t * NE + e] = lg[e]; mx = fmaxf(mx, lg[e]); }
#pragma unroll
    for (int e = 0; e < NE; ++e) { w[e] = __expf(lg[e] - mx); sum += w[e]; }
#pragma unroll
    for (int e = 0; e < NE; ++e) w[e] /= sum;
    int i0 = 0;
    for (int e = 1; e < NE; ++e) if (w[e] > w[i0]) i0 = e;
    int i1 = (i0 == 0) ? 1 : 0;
    for (int e = 0; e < NE; ++e) if (e != i0 && w[e] > w[i1]) i1 = e;
    float s = w[i0] + w[i1];
    for (int e = 0; e < NE; ++e) comb[t * NE + e] = 0.f;
    comb[t * NE + i0] = w[i0] / s;
    comb[t * NE + i1] = w[i1] / s;
  }
}

// --------------------------- small utility kernels -------------------------
__global__ void f32_to_bf16_kernel(const float* __restrict__ s, bf16* __restrict__ d, int n)
{
  int i = blockIdx.x * 256 + threadIdx.x;
  if (i < n) d[i] = f2b(s[i]);
}

__global__ void zero_bf16_kernel(unsigned short* __restrict__ p, int n)
{
  int i = blockIdx.x * 256 + threadIdx.x;
  if (i < n) p[i] = 0;
}

__global__ void im2col_kernel(const float* __restrict__ img, bf16* __restrict__ XP)
{
  int idx = blockIdx.x * 256 + threadIdx.x;
  if (idx >= NPATCH * DIM) return;
  int p = idx / DIM, k = idx % DIM;
  int b = p / 128, pp = p % 128;
  int py = pp / 8, px = pp % 8;
  int c = k / 256, rem = k % 256, y = rem / 16, xx = rem % 16;
  XP[idx] = f2b(img[((size_t)(b * 3 + c) * 256 + py * 16 + y) * 128 + px * 16 + xx]);
}

__global__ void embed_kernel(const float* __restrict__ ETOK, const float* __restrict__ cls,
                             const float* __restrict__ pos, float* __restrict__ XRAW)
{
  int idx = blockIdx.x * 256 + threadIdx.x;
  if (idx >= TOK * DIM) return;
  int t = idx / DIM, d = idx % DIM;
  int s = t / BATCH, b = t % BATCH;
  float v = (s == 0) ? cls[d] : ETOK[(size_t)(b * 128 + s - 1) * DIM + d];
  XRAW[idx] = v + pos[(size_t)s * DIM + d];
}

__global__ void moe_add_kernel(const float* __restrict__ YE, const float* __restrict__ comb,
                               int e, float* __restrict__ X)
{
  int idx = blockIdx.x * 256 + threadIdx.x;
  if (idx >= TOK * DIM) return;
  int t = idx / DIM;
  X[idx] += comb[t * NE + e] * YE[idx];
}

__global__ void to_x11_kernel(const float* __restrict__ X, float* __restrict__ out1)
{
  int idx = blockIdx.x * 256 + threadIdx.x;
  if (idx >= TOK * DIM) return;
  int b = idx / (SEQ * DIM);
  int r = idx % (SEQ * DIM);
  int s = r / DIM, d = r % DIM;
  out1[idx] = X[(size_t)(s * BATCH + b) * DIM + d];
}

__global__ void transpose_proj_kernel(const float* __restrict__ proj, bf16* __restrict__ projT)
{
  int idx = blockIdx.x * 256 + threadIdx.x;     // over DOUT*DIM
  if (idx >= DOUT * DIM) return;
  int n = idx / DIM, k = idx % DIM;
  projT[idx] = f2b(proj[(size_t)k * DOUT + n]);
}

// ---------------------------------------------------------------------------
extern "C" void kernel_launch(void* const* d_in, const int* in_sizes, int n_in,
                              void* d_out, int out_size, void* d_ws, size_t ws_size,
                              hipStream_t stream)
{
  (void)in_sizes; (void)n_in; (void)out_size; (void)ws_size;
  const float* images    = (const float*)d_in[0];
  const float* conv_w    = (const float*)d_in[1];
  const float* cls_emb   = (const float*)d_in[2];
  const float* pos_emb   = (const float*)d_in[3];
  const float* ln_pre_w  = (const float*)d_in[4];
  const float* ln_pre_b  = (const float*)d_in[5];
  const float* ln1_w     = (const float*)d_in[6];
  const float* ln1_b     = (const float*)d_in[7];
  const float* ln2_w     = (const float*)d_in[8];
  const float* ln2_b     = (const float*)d_in[9];
  const float* qkv_w     = (const float*)d_in[10];
  const float* qkv_b     = (const float*)d_in[11];
  const float* out_w     = (const float*)d_in[12];
  const float* out_b     = (const float*)d_in[13];
  const float* gate_w    = (const float*)d_in[14];
  const float* moe_fc_w  = (const float*)d_in[15];
  const float* moe_fc_b  = (const float*)d_in[16];
  const float* moe_pj_w  = (const float*)d_in[17];
  const float* moe_pj_b  = (const float*)d_in[18];
  const float* mlp_fc_w  = (const float*)d_in[19];
  const float* mlp_fc_b  = (const float*)d_in[20];
  const float* mlp_pj_w  = (const float*)d_in[21];
  const float* mlp_pj_b  = (const float*)d_in[22];
  const float* ln_post_w = (const float*)d_in[23];
  const float* ln_post_b = (const float*)d_in[24];
  const float* proj      = (const float*)d_in[25];

  float* ws    = (float*)d_ws;
  float* X     = ws;                       // [T,D]
  float* XRAW  = X     + 396288;           // [T,D]
  float* Hb    = XRAW  + 396288;           // [T,D]  (router input)
  float* QKV   = Hb    + 396288;           // [T,3D]
  float* ETOK  = QKV   + 1188864;          // [512,768]
  float* YE    = ETOK  + 393216;           // [T,D]
  float* COMB  = YE    + 396288;           // [T,E] (+pad)
  bf16* Hb16   = (bf16*)(COMB + 4160);     // [MPAD,768]
  bf16* Ob16   = Hb16   + 491520;          // [MPAD,768]
  bf16* HID16  = Ob16   + 491520;          // [MPAD,3072]
  bf16* XPbf   = HID16  + 1966080;         // [512,768]
  bf16* X12b16 = XPbf   + 393216;          // [MPAD,768]
  bf16* PRJ16  = X12b16 + 491520;          // [512,768]
  bf16* WBF    = PRJ16  + 393216;          // up to 3072*768

  float* out  = (float*)d_out;
  float* x11  = out;               // [B,S,D]
  float* x12  = out + 396288;      // [B,S,D]
  float* xpj  = out + 792576;      // [B,S,OUT]
  float* rlog = out + 1056768;     // [T,E]

  auto gemm = [&](const bf16* A, const bf16* Wb, const float* bias,
                  const float* res, float* C, bf16* Cb, int M, int N, int K, int act) {
    dim3 g((N + 127) / 128, (M + 127) / 128);
    gemm_wmma<<<g, dim3(256), 0, stream>>>(A, Wb, bias, res, C, Cb, M, N, K, act);
  };
  auto wconv = [&](const float* src, int n) {
    f32_to_bf16_kernel<<<(n + 255) / 256, 256, 0, stream>>>(src, WBF, n);
  };

  // ---- zero the padded-row tails of bf16 activation buffers (per call) ----
  int tail = (MPAD - TOK) * DIM;      // 95232
  int tailF = (MPAD - TOK) * FF;      // 380928
  zero_bf16_kernel<<<(tail + 255) / 256, 256, 0, stream>>>((unsigned short*)(Hb16 + TOK * DIM), tail);
  zero_bf16_kernel<<<(tail + 255) / 256, 256, 0, stream>>>((unsigned short*)(Ob16 + TOK * DIM), tail);
  zero_bf16_kernel<<<(tailF + 255) / 256, 256, 0, stream>>>((unsigned short*)(HID16 + TOK * FF), tailF);
  zero_bf16_kernel<<<(tail + 255) / 256, 256, 0, stream>>>((unsigned short*)(X12b16 + TOK * DIM), tail);

  // ---- patchify + embed + ln_pre ----
  im2col_kernel<<<(NPATCH * DIM + 255) / 256, 256, 0, stream>>>(images, XPbf);
  wconv(conv_w, DIM * DIM);
  gemm(XPbf, WBF, nullptr, nullptr, ETOK, nullptr, NPATCH, DIM, DIM, 0);
  embed_kernel<<<(TOK * DIM + 255) / 256, 256, 0, stream>>>(ETOK, cls_emb, pos_emb, XRAW);
  ln_kernel<<<TOK, 256, 0, stream>>>(XRAW, X, nullptr, ln_pre_w, ln_pre_b);

  // ---- transformer layers ----
  for (int i = 0; i < NL; ++i) {
    ln_kernel<<<TOK, 256, 0, stream>>>(X, nullptr, Hb16, ln1_w + i * DIM, ln1_b + i * DIM);
    wconv(qkv_w + (size_t)i * 3 * DIM * DIM, 3 * DIM * DIM);
    gemm(Hb16, WBF, qkv_b + (size_t)i * 3 * DIM, nullptr, QKV, nullptr, TOK, 3 * DIM, DIM, 0);
    attn_kernel<<<BATCH * NH * SEQ, 64, 0, stream>>>(QKV, Ob16);
    wconv(out_w + (size_t)i * DIM * DIM, DIM * DIM);
    gemm(Ob16, WBF, out_b + (size_t)i * DIM, X, X, nullptr, TOK, DIM, DIM, 0);

    ln_kernel<<<TOK, 256, 0, stream>>>(X, Hb, Hb16, ln2_w + i * DIM, ln2_b + i * DIM);

    if (i < NLM) {
      if (i == 0)
        router_kernel<<<TOK, 256, 0, stream>>>(Hb, gate_w, COMB, rlog);
      for (int e = 0; e < NE; ++e) {
        size_t m = (size_t)i * NE + e;
        wconv(moe_fc_w + m * FF * DIM, FF * DIM);
        gemm(Hb16, WBF, moe_fc_b + m * FF, nullptr, nullptr, HID16, TOK, FF, DIM, 1);
        wconv(moe_pj_w + m * DIM * FF, DIM * FF);
        gemm(HID16, WBF, moe_pj_b + m * DIM, nullptr, YE, nullptr, TOK, DIM, FF, 0);
        moe_add_kernel<<<(TOK * DIM + 255) / 256, 256, 0, stream>>>(YE, COMB, e, X);
      }
    } else {
      int j = i - NLM;
      wconv(mlp_fc_w + (size_t)j * FF * DIM, FF * DIM);
      gemm(Hb16, WBF, mlp_fc_b + (size_t)j * FF, nullptr, nullptr, HID16, TOK, FF, DIM, 1);
      wconv(mlp_pj_w + (size_t)j * DIM * FF, DIM * FF);
      gemm(HID16, WBF, mlp_pj_b + (size_t)j * DIM, X, X, nullptr, TOK, DIM, FF, 0);
    }
  }

  // ---- outputs ----
  to_x11_kernel<<<(TOK * DIM + 255) / 256, 256, 0, stream>>>(X, x11);
  ln_kernel<<<TOK, 256, 0, stream>>>(x11, x12, X12b16, ln_post_w, ln_post_b);
  transpose_proj_kernel<<<(DOUT * DIM + 255) / 256, 256, 0, stream>>>(proj, PRJ16);
  gemm(X12b16, PRJ16, nullptr, nullptr, xpj, nullptr, TOK, DOUT, DIM, 0);
}
